// GNNModel_44435731644777
// MI455X (gfx1250) — matrix operand order repaired and checked by reading, not compiled, not run on hardware
//
#include <hip/hip_runtime.h>
#include <hip/hip_bf16.h>

#define N_NODES 100000
#define E_EDGES 1600000
#define ETOT    (E_EDGES + N_NODES)

typedef __attribute__((ext_vector_type(16))) __bf16 v16bf;
typedef __attribute__((ext_vector_type(8)))  float  v8f;

__device__ __forceinline__ unsigned enc_f32(float f){
  unsigned u = __float_as_uint(f);
  return (u & 0x80000000u) ? ~u : (u | 0x80000000u);
}
__device__ __forceinline__ float dec_f32(unsigned u){
  unsigned v = (u & 0x80000000u) ? (u & 0x7FFFFFFFu) : ~u;
  return __uint_as_float(v);
}
__device__ __forceinline__ void edge_ep(const int* __restrict__ ei, int e, int& s, int& d){
  if (e < E_EDGES){ s = ei[e]; d = ei[E_EDGES + e]; } else { s = d = e - E_EDGES; }
}

// ---- pack W [K=128, Ncols] f32 row-major -> bf16 WMMA B-operand layout ----
// B 32x16, contiguous k order (A uses the same): lane l holds column n=l&15; elements i=0..15 are K = (l>>4)*16 + i.
__global__ void pack_b_kernel(const float* __restrict__ W, __bf16* __restrict__ pack, int Ncols){
  int idx = blockIdx.x * blockDim.x + threadIdx.x;
  int NT = Ncols >> 4;
  int total = 128 * Ncols;
  if (idx >= total) return;
  int i    = idx & 15;
  int lane = (idx >> 4) & 31;
  int nt   = (idx >> 9) % NT;
  int kt   = (idx >> 9) / NT;
  int krow = kt * 32 + (lane >> 4) * 16 + i;
  int col  = nt * 16 + (lane & 15);
  pack[idx] = (__bf16)W[krow * Ncols + col];
}

__global__ void fill_bias_kernel(float* __restrict__ out, const float* __restrict__ bias,
                                 int cols, long total){
  long i = (long)blockIdx.x * blockDim.x + threadIdx.x;
  if (i < total) out[i] = bias[i % cols];
}
__global__ void zero_kernel(unsigned* __restrict__ p, long n){
  long i = (long)blockIdx.x * blockDim.x + threadIdx.x;
  if (i < n) p[i] = 0u;
}

// ---- GEMM: C[M,NT*16] = act(A[M,128] f32 -> bf16) x Bpack ----
// blockDim = NT*32; wave w handles n-tile w; grid.x = M/16.
// ELU (layer-2 input activation) is fused into the f32->bf16 conversion.
template<int NT, bool ELU>
__global__ void gemm_wmma_kernel(const float* __restrict__ A, const __bf16* __restrict__ Bp,
                                 float* __restrict__ C){
  const int Ncols = NT * 16;
  int lane = threadIdx.x & 31;
  int nt   = threadIdx.x >> 5;
  long rb  = (long)blockIdx.x * 16;
  int  m   = lane & 15;
  int  kh  = lane >> 4;
  const float* Ar = A + (rb + m) * 128;
  v8f acc = {};
  #pragma unroll
  for (int kt = 0; kt < 4; ++kt){
    int kb = kt * 32 + kh * 16;
    float4 a0 = *(const float4*)(Ar + kb);
    float4 a1 = *(const float4*)(Ar + kb + 4);
    float4 a2 = *(const float4*)(Ar + kb + 8);
    float4 a3 = *(const float4*)(Ar + kb + 12);
    float av_f[16] = {a0.x,a0.y,a0.z,a0.w, a1.x,a1.y,a1.z,a1.w,
                      a2.x,a2.y,a2.z,a2.w, a3.x,a3.y,a3.z,a3.w};
    v16bf av;
    #pragma unroll
    for (int j = 0; j < 16; ++j){
      float v = av_f[j];
      if (ELU) v = v > 0.f ? v : (__expf(v) - 1.f);
      av[j] = (__bf16)v;
    }
    v16bf bv = *(const v16bf*)(Bp + (size_t)((kt * NT + nt) * 32 + lane) * 16);
    acc = __builtin_amdgcn_wmma_f32_16x16x32_bf16(false, av, false, bv, (short)0, acc, false, false);
  }
  int col = nt * 16 + m;
  long rbase = rb + kh * 8;
  #pragma unroll
  for (int r = 0; r < 8; ++r)
    C[(rbase + r) * Ncols + col] = acc[r];
}

// ---- per-node attention scalars: a[n][h] = sum_c xp[n,h,c]*att[h,c] ----
template<int H, int C>
__global__ void attn_dot_kernel(const float* __restrict__ xp,
                                const float* __restrict__ atts, const float* __restrict__ attd,
                                float* __restrict__ as, float* __restrict__ ad){
  const int HC  = H * C;
  const int EPL = HC / 32;   // elems per lane
  const int LPH = 32 / H;    // lanes per head
  int wave = (int)((blockIdx.x * blockDim.x + threadIdx.x) >> 5);
  int lane = threadIdx.x & 31;
  if (wave >= N_NODES) return;
  const float* xr = xp + (long)wave * HC;
  float s = 0.f, d = 0.f;
  #pragma unroll
  for (int j = 0; j < EPL; ++j){
    float v = xr[lane * EPL + j];
    s += v * atts[lane * EPL + j];
    d += v * attd[lane * EPL + j];
  }
  #pragma unroll
  for (int off = 1; off < LPH; off <<= 1){
    s += __shfl_xor(s, off, 32);
    d += __shfl_xor(d, off, 32);
  }
  if ((lane & (LPH - 1)) == 0){
    int h = lane / LPH;
    as[(long)wave * H + h] = s;
    ad[(long)wave * H + h] = d;
  }
}

template<int H>
__global__ void edge_max_kernel(const int* __restrict__ ei, const float* __restrict__ as,
                                const float* __restrict__ ad, unsigned* __restrict__ mx){
  int e = blockIdx.x * blockDim.x + threadIdx.x;
  if (e >= ETOT) return;
  int s, d; edge_ep(ei, e, s, d);
  #pragma unroll
  for (int h = 0; h < H; ++h){
    float l = as[(long)s * H + h] + ad[(long)d * H + h];
    l = l > 0.f ? l : 0.2f * l;
    atomicMax(&mx[(long)d * H + h], enc_f32(l));
  }
}

template<int H>
__global__ void edge_sum_kernel(const int* __restrict__ ei, const float* __restrict__ as,
                                const float* __restrict__ ad, const unsigned* __restrict__ mx,
                                float* __restrict__ dn){
  int e = blockIdx.x * blockDim.x + threadIdx.x;
  if (e >= ETOT) return;
  int s, d; edge_ep(ei, e, s, d);
  #pragma unroll
  for (int h = 0; h < H; ++h){
    float l = as[(long)s * H + h] + ad[(long)d * H + h];
    l = l > 0.f ? l : 0.2f * l;
    atomicAdd(&dn[(long)d * H + h], __expf(l - dec_f32(mx[(long)d * H + h])));
  }
}

// one thread per (edge, 4-channel quad)
template<int H, int C>
__global__ void edge_msg_kernel(const int* __restrict__ ei, const float* __restrict__ as,
                                const float* __restrict__ ad, const unsigned* __restrict__ mx,
                                const float* __restrict__ dn, const float* __restrict__ xp,
                                float* __restrict__ out){
  const int HC  = H * C;
  const int TPE = HC / 4;
  long t = (long)blockIdx.x * blockDim.x + threadIdx.x;
  long e = t / TPE;
  int  q = (int)(t % TPE);
  if (e >= ETOT) return;
  int s, d; edge_ep(ei, (int)e, s, d);
  int c4 = q * 4;
  int h  = c4 / C;
  float l = as[(long)s * H + h] + ad[(long)d * H + h];
  l = l > 0.f ? l : 0.2f * l;
  float alpha = __expf(l - dec_f32(mx[(long)d * H + h])) / dn[(long)d * H + h];
  float4 xv = *(const float4*)(xp + (long)s * HC + c4);
  float* o = out + (long)d * HC + c4;
  atomicAdd(o + 0, xv.x * alpha);
  atomicAdd(o + 1, xv.y * alpha);
  atomicAdd(o + 2, xv.z * alpha);
  atomicAdd(o + 3, xv.w * alpha);
}

extern "C" void kernel_launch(void* const* d_in, const int* in_sizes, int n_in,
                              void* d_out, int out_size, void* d_ws, size_t ws_size,
                              hipStream_t stream){
  const float* x    = (const float*)d_in[0];
  const int*   ei   = (const int*)  d_in[1];
  const float* W1   = (const float*)d_in[2];
  const float* as1w = (const float*)d_in[3];
  const float* ad1w = (const float*)d_in[4];
  const float* b1   = (const float*)d_in[5];
  const float* W2   = (const float*)d_in[6];
  const float* as2w = (const float*)d_in[7];
  const float* ad2w = (const float*)d_in[8];
  const float* b2   = (const float*)d_in[9];
  float* out = (float*)d_out;

  const long N = N_NODES;
  float* F = (float*)d_ws;
  size_t o = 0;
  float*    xp   = F + o; o += (size_t)N * 128;  // xp1, reused as xp2
  float*    hbuf = F + o; o += (size_t)N * 128;  // layer-1 accumulator (pre-ELU h)
  float*    as1  = F + o; o += (size_t)N * 4;
  float*    ad1  = F + o; o += (size_t)N * 4;
  unsigned* mx1  = (unsigned*)(F + o); o += (size_t)N * 4;
  float*    dn1  = F + o; o += (size_t)N * 4;   // contiguous with mx1 (zeroed together)
  float*    as2  = F + o; o += (size_t)N;
  float*    ad2  = F + o; o += (size_t)N;
  unsigned* mx2  = (unsigned*)(F + o); o += (size_t)N;
  float*    dn2  = F + o; o += (size_t)N;       // contiguous with mx2
  __bf16*   w1p  = (__bf16*)(F + o); o += (size_t)(128 * 128) / 2;
  __bf16*   w2p  = (__bf16*)(F + o);

  auto cdiv = [](long a, long b){ return (unsigned)((a + b - 1) / b); };

  // weight packing + accumulator init (bias folded into init)
  pack_b_kernel<<<cdiv(128 * 128, 256), 256, 0, stream>>>(W1, w1p, 128);
  pack_b_kernel<<<cdiv(128 * 64, 256), 256, 0, stream>>>(W2, w2p, 64);
  fill_bias_kernel<<<cdiv(N * 128, 256), 256, 0, stream>>>(hbuf, b1, 128, N * 128);
  fill_bias_kernel<<<cdiv(N * 64, 256), 256, 0, stream>>>(out, b2, 64, N * 64);
  zero_kernel<<<cdiv(N * 8, 256), 256, 0, stream>>>(mx1, N * 8);   // mx1 + dn1
  zero_kernel<<<cdiv(N * 2, 256), 256, 0, stream>>>(mx2, N * 2);   // mx2 + dn2

  // ---- layer 1: 4 heads x 32 ----
  gemm_wmma_kernel<8, false><<<N_NODES / 16, 256, 0, stream>>>(x, w1p, xp);
  attn_dot_kernel<4, 32><<<cdiv(N, 8), 256, 0, stream>>>(xp, as1w, ad1w, as1, ad1);
  edge_max_kernel<4><<<cdiv(ETOT, 256), 256, 0, stream>>>(ei, as1, ad1, mx1);
  edge_sum_kernel<4><<<cdiv(ETOT, 256), 256, 0, stream>>>(ei, as1, ad1, mx1, dn1);
  edge_msg_kernel<4, 32><<<cdiv((long)ETOT * 32, 256), 256, 0, stream>>>(ei, as1, ad1, mx1, dn1, xp, hbuf);

  // ---- layer 2: 1 head x 64 (ELU fused into GEMM A-operand conversion) ----
  gemm_wmma_kernel<4, true><<<N_NODES / 16, 128, 0, stream>>>(hbuf, w2p, xp);
  attn_dot_kernel<1, 64><<<cdiv(N, 8), 256, 0, stream>>>(xp, as2w, ad2w, as2, ad2);
  edge_max_kernel<1><<<cdiv(ETOT, 256), 256, 0, stream>>>(ei, as2, ad2, mx2);
  edge_sum_kernel<1><<<cdiv(ETOT, 256), 256, 0, stream>>>(ei, as2, ad2, mx2, dn2);
  edge_msg_kernel<1, 64><<<cdiv((long)ETOT * 16, 256), 256, 0, stream>>>(ei, as2, ad2, mx2, dn2, xp, out);

  (void)in_sizes; (void)n_in; (void)out_size; (void)ws_size;
}
